// LinearCRF_738734375058
// MI455X (gfx1250) — compile-verified, matrix operand order checked by reading, not run on hardware
//
#include <hip/hip_runtime.h>
#include <hip/hip_bf16.h>

#define S_DIM 512
#define B_DIM 64
#define H_DIM 512
#define T_DIM 32
#define M_DIM (S_DIM * B_DIM)      // 32768 rows
#define KT    (H_DIM / 32)         // 16 K-tiles (K=32 per WMMA)
#define NT1   (H_DIM / 16)         // 32 N-tiles for GEMM1
#define NT2   (T_DIM / 16)         // 2  N-tiles for GEMM2
#define WAVES 4
#define MROWS 32                   // rows per wave (2 x 16-row M-tiles)

typedef __attribute__((ext_vector_type(16))) __bf16 v16bf;
typedef __attribute__((ext_vector_type(8)))  __bf16 v8bf;
typedef __attribute__((ext_vector_type(8)))  float  v8f;
typedef __attribute__((ext_vector_type(4)))  float  v4f;
typedef __attribute__((ext_vector_type(4)))  int    v4i;

typedef __attribute__((address_space(1))) v4i GV4i;   // global b128 payload
typedef __attribute__((address_space(3))) v4i LV4i;   // LDS b128 payload

// Workspace layout:
//   [0)                : W1 bf16, B-operand swizzled tiles (KT*NT1 tiles, 1KB each)
//   [+512KB)           : W2 bf16, B-operand swizzled tiles
//   [WS_EMIS_BYTE_OFF) : emissions f32 [M][T]  (4 MB)
#define WS_W1_ELEMS (KT * NT1 * 32 * 16)
#define WS_W2_ELEMS (KT * NT2 * 32 * 16)
#define WS_EMIS_BYTE_OFF ((size_t)(WS_W1_ELEMS + WS_W2_ELEMS) * 2)

// ---------------------------------------------------------------------------
// Async-copy helpers (gfx1250): 16B global -> LDS, ASYNCcnt-tracked.
// ---------------------------------------------------------------------------
__device__ __forceinline__ void cp_async_b128(const void* g, void* l) {
#if __has_builtin(__builtin_amdgcn_global_load_async_to_lds_b128)
    __builtin_amdgcn_global_load_async_to_lds_b128((GV4i*)g, (LV4i*)l, 0, 0);
#else
    unsigned laddr = (unsigned)(size_t)(__attribute__((address_space(3))) char*)l;
    unsigned long long gaddr = (unsigned long long)(size_t)g;
    asm volatile("global_load_async_to_lds_b128 %0, %1, off"
                 :: "v"(laddr), "v"(gaddr) : "memory");
#endif
}

__device__ __forceinline__ void wait_async0() {
#if __has_builtin(__builtin_amdgcn_s_wait_asynccnt)
    __builtin_amdgcn_s_wait_asynccnt(0);
#else
    asm volatile("s_wait_asynccnt 0x0" ::: "memory");
#endif
}

// ---------------------------------------------------------------------------
// Kernel 1: convert W1/W2 (f32 row-major [K][N]) into bf16 WMMA B-tile layout.
//   tile (kt,nt): lane L holds 16 contiguous bf16:
//   col = nt*16 + (L&15), k0 = kt*32 + (L<16?0:8); e0..7=K[k0..], e8..15=K[k0+16..]
// ---------------------------------------------------------------------------
__global__ void swizzle_weights(const float* __restrict__ W1,
                                const float* __restrict__ W2,
                                __bf16* __restrict__ w1s,
                                __bf16* __restrict__ w2s) {
    const int idx = blockIdx.x * blockDim.x + threadIdx.x;
    const int totalW1 = KT * NT1 * 32;
    if (idx < totalW1) {
        const int lane = idx & 31;
        const int tile = idx >> 5;
        const int nt = tile % NT1;
        const int kt = tile / NT1;
        const int col = nt * 16 + (lane & 15);
        const int k0  = kt * 32 + ((lane < 16) ? 0 : 8);
        __bf16* out = w1s + (size_t)idx * 16;
#pragma unroll
        for (int e = 0; e < 8; ++e) {
            out[e]     = (__bf16)W1[(size_t)(k0 + e) * H_DIM + col];
            out[8 + e] = (__bf16)W1[(size_t)(k0 + 16 + e) * H_DIM + col];
        }
    } else {
        const int j = idx - totalW1;
        if (j < KT * NT2 * 32) {
            const int lane = j & 31;
            const int tile = j >> 5;
            const int nt = tile % NT2;
            const int kt = tile / NT2;
            const int col = nt * 16 + (lane & 15);
            const int k0  = kt * 32 + ((lane < 16) ? 0 : 8);
            __bf16* out = w2s + (size_t)j * 16;
#pragma unroll
            for (int e = 0; e < 8; ++e) {
                out[e]     = (__bf16)W2[(size_t)(k0 + e) * T_DIM + col];
                out[8 + e] = (__bf16)W2[(size_t)(k0 + 16 + e) * T_DIM + col];
            }
        }
    }
}

// ---------------------------------------------------------------------------
// Kernel 2: emissions = (sigmoid(hidden @ W1 + b1)) @ W2 + b2  via bf16 WMMA.
// Per wave: 32-row M-tile (A in 256 VGPRs, 2 accumulators per B-tile).
// W1 B-tiles double-buffered in LDS via async copies, shared by all 4 waves.
// ---------------------------------------------------------------------------
__global__ __launch_bounds__(WAVES * 32)
void emissions_kernel(const float* __restrict__ hidden,
                      const float* __restrict__ b1,
                      const float* __restrict__ b2,
                      const __bf16* __restrict__ w1s,
                      const __bf16* __restrict__ w2s,
                      float* __restrict__ emis) {
    __shared__ __bf16 h1[WAVES][MROWS][H_DIM];   // 128 KB
    __shared__ __bf16 bB[2][KT][32 * 16];        // 2 x 16 KB double buffer

    const int tid   = threadIdx.x;
    const int lane  = tid & 31;
    const int w     = tid >> 5;
    const int half  = lane >> 4;                 // 0: K lo-run, 1: K hi-run
    const int rrow  = lane & 15;                 // A row / B,C column within tile
    const int m0    = (blockIdx.x * WAVES + w) * MROWS;

    // --- Preload A: 32 rows x 512 K of `hidden`, f32 -> bf16, in registers --
    v16bf a0[KT], a1[KT];
    const float* rp0 = hidden + (size_t)(m0 + rrow) * H_DIM;
    const float* rp1 = hidden + (size_t)(m0 + 16 + rrow) * H_DIM;
#pragma unroll
    for (int kt = 0; kt < KT; ++kt) {
        const int k0 = kt * 32 + half * 8;
        v4f f0 = *(const v4f*)(rp0 + k0);
        v4f f1 = *(const v4f*)(rp0 + k0 + 4);
        v4f f2 = *(const v4f*)(rp0 + k0 + 16);
        v4f f3 = *(const v4f*)(rp0 + k0 + 20);
        v4f g0 = *(const v4f*)(rp1 + k0);
        v4f g1 = *(const v4f*)(rp1 + k0 + 4);
        v4f g2 = *(const v4f*)(rp1 + k0 + 16);
        v4f g3 = *(const v4f*)(rp1 + k0 + 20);
#pragma unroll
        for (int e = 0; e < 4; ++e) {
            a0[kt][e]      = (__bf16)f0[e];
            a0[kt][4 + e]  = (__bf16)f1[e];
            a0[kt][8 + e]  = (__bf16)f2[e];
            a0[kt][12 + e] = (__bf16)f3[e];
            a1[kt][e]      = (__bf16)g0[e];
            a1[kt][4 + e]  = (__bf16)g1[e];
            a1[kt][8 + e]  = (__bf16)g2[e];
            a1[kt][12 + e] = (__bf16)g3[e];
        }
    }

    // Async prefetch of one N-tile's 16 B-tiles (16 KB) into LDS buffer p.
    // 128 threads x 128 B each: kt = tid>>3, intra-tile base = (tid&7)*128.
    const char* w1bytes = (const char*)w1s;
    auto prefetchB = [&](int p, int nt) {
        const int kt   = tid >> 3;
        const int base = (tid & 7) * 128;
        const char* gsrc = w1bytes + ((size_t)(kt * NT1 + nt) << 10) + base;
        char* ldst = (char*)&bB[p][kt][0] + base;
#pragma unroll
        for (int j = 0; j < 8; ++j)
            cp_async_b128(gsrc + j * 16, ldst + j * 16);
    };

    prefetchB(0, 0);

    // --- GEMM1: h1 = sigmoid(hidden @ W1 + b1), bf16 -> LDS ----------------
#pragma unroll 1
    for (int nt = 0; nt < NT1; ++nt) {
        const int p = nt & 1;
        wait_async0();           // my slice of buffer p landed
        __syncthreads();         // everyone's slice landed; prior compute done
        if (nt + 1 < NT1) prefetchB(p ^ 1, nt + 1);

        v8f acc0 = {}, acc1 = {};
#pragma unroll
        for (int kt = 0; kt < KT; ++kt) {
            v16bf b = *(const v16bf*)&bB[p][kt][lane * 16];
            acc0 = __builtin_amdgcn_wmma_f32_16x16x32_bf16(
                false, a0[kt], false, b, (short)0, acc0, false, false);
            acc1 = __builtin_amdgcn_wmma_f32_16x16x32_bf16(
                false, a1[kt], false, b, (short)0, acc1, false, false);
        }
        const int col   = nt * 16 + rrow;
        const float bia = b1[col];
#pragma unroll
        for (int r = 0; r < 8; ++r) {
            float x0 = acc0[r] + bia;
            float x1 = acc1[r] + bia;
            h1[w][r + half * 8][col]      = (__bf16)(1.0f / (1.0f + __expf(-x0)));
            h1[w][16 + r + half * 8][col] = (__bf16)(1.0f / (1.0f + __expf(-x1)));
        }
        __syncthreads();         // buffer p may be overwritten next iteration
    }

    // --- GEMM2: emissions = h1 @ W2 + b2 -----------------------------------
    const v16bf* w2t = (const v16bf*)w2s;
#pragma unroll 1
    for (int nt = 0; nt < NT2; ++nt) {
        v8f acc0 = {}, acc1 = {};
#pragma unroll
        for (int kt = 0; kt < KT; ++kt) {
            const int k0 = kt * 32 + half * 8;
            v16bf a2, a3;
            v8bf lo0 = *(const v8bf*)&h1[w][rrow][k0];
            v8bf hi0 = *(const v8bf*)&h1[w][rrow][k0 + 16];
            v8bf lo1 = *(const v8bf*)&h1[w][16 + rrow][k0];
            v8bf hi1 = *(const v8bf*)&h1[w][16 + rrow][k0 + 16];
#pragma unroll
            for (int e = 0; e < 8; ++e) {
                a2[e] = lo0[e]; a2[8 + e] = hi0[e];
                a3[e] = lo1[e]; a3[8 + e] = hi1[e];
            }
            v16bf b = w2t[(size_t)(kt * NT2 + nt) * 32 + lane];
            acc0 = __builtin_amdgcn_wmma_f32_16x16x32_bf16(
                false, a2, false, b, (short)0, acc0, false, false);
            acc1 = __builtin_amdgcn_wmma_f32_16x16x32_bf16(
                false, a3, false, b, (short)0, acc1, false, false);
        }
        const int col   = nt * 16 + rrow;
        const float bia = b2[col];
#pragma unroll
        for (int r = 0; r < 8; ++r) {
            const int ma = m0 + r + half * 8;
            const int mb = m0 + 16 + r + half * 8;
            emis[(size_t)ma * T_DIM + col] = acc0[r] + bia;
            emis[(size_t)mb * T_DIM + col] = acc1[r] + bia;
        }
    }
}

// ---------------------------------------------------------------------------
// Kernel 3: Viterbi forward + backtrace. One wave32 per batch element;
// lane == current tag. Backpointers live entirely in LDS (16 KB).
// ---------------------------------------------------------------------------
__global__ __launch_bounds__(32)
void viterbi_kernel(const float* __restrict__ emis,
                    const int* __restrict__ lens,
                    const float* __restrict__ transition,
                    int* __restrict__ paths,    // [S][B]
                    float* __restrict__ score)  // [B]
{
    __shared__ float fv_s[T_DIM];
    __shared__ unsigned char bp[S_DIM][T_DIM];   // slot t in [1, S)

    const int b    = blockIdx.x;
    const int lane = threadIdx.x;                // tag 'curr'
    const int len  = lens[b];

    float trow[T_DIM];                           // transition[curr=lane][prev]
#pragma unroll
    for (int p = 0; p < T_DIM; ++p) trow[p] = transition[lane * T_DIM + p];

    float fv = emis[(size_t)b * T_DIM + lane];   // emissions[0][b][lane]

    for (int t = 1; t < S_DIM; ++t) {
        fv_s[lane] = fv;
        __syncthreads();
        float best = -3.402823466e38f;
        int   arg  = 0;
#pragma unroll
        for (int p = 0; p < T_DIM; ++p) {
            float sc = fv_s[p] + trow[p];
            if (sc > best) { best = sc; arg = p; }   // strict > : first max (jnp.argmax)
        }
        const bool active = (t < len);
        const float em = emis[((size_t)t * B_DIM + b) * T_DIM + lane];
        fv = active ? (best + em) : fv;
        bp[t][lane] = (unsigned char)(active ? arg : lane);  // identity when frozen
        __syncthreads();
    }

    fv_s[lane] = fv;
    __syncthreads();
    if (lane == 0) {
        float best = fv_s[0];
        int   bid  = 0;
#pragma unroll
        for (int p = 1; p < T_DIM; ++p) {
            if (fv_s[p] > best) { best = fv_s[p]; bid = p; }
        }
        score[b] = best;
        paths[(size_t)(S_DIM - 1) * B_DIM + b] = bid;
        int tag = bid;
        for (int t = S_DIM - 1; t >= 1; --t) {
            const int prev = bp[t][tag];
            paths[(size_t)(t - 1) * B_DIM + b] = prev;
            tag = prev;
        }
    }
}

// ---------------------------------------------------------------------------
extern "C" void kernel_launch(void* const* d_in, const int* in_sizes, int n_in,
                              void* d_out, int out_size, void* d_ws, size_t ws_size,
                              hipStream_t stream) {
    (void)in_sizes; (void)n_in; (void)out_size; (void)ws_size;
    const float* hidden     = (const float*)d_in[0];
    const int*   lens       = (const int*)d_in[1];
    /* d_in[2] = masks (unused by reference) */
    const float* W1         = (const float*)d_in[3];
    const float* b1         = (const float*)d_in[4];
    const float* W2         = (const float*)d_in[5];
    const float* b2         = (const float*)d_in[6];
    const float* transition = (const float*)d_in[7];

    __bf16* w1s = (__bf16*)d_ws;
    __bf16* w2s = w1s + WS_W1_ELEMS;
    float*  emis = (float*)((char*)d_ws + WS_EMIS_BYTE_OFF);

    const int convThreads = KT * NT1 * 32 + KT * NT2 * 32;
    swizzle_weights<<<(convThreads + 255) / 256, 256, 0, stream>>>(W1, W2, w1s, w2s);

    emissions_kernel<<<M_DIM / (MROWS * WAVES), WAVES * 32, 0, stream>>>(
        hidden, b1, b2, w1s, w2s, emis);

    int*   paths = (int*)d_out;                            // [S][B] int32
    float* scr   = (float*)d_out + (size_t)S_DIM * B_DIM;  // [B] f32
    viterbi_kernel<<<B_DIM, 32, 0, stream>>>(emis, lens, transition, paths, scr);
}
